// EnhancedFusionModel_19061064860226
// MI455X (gfx1250) — matrix operand order repaired and checked by reading, not compiled, not run on hardware
//
#include <hip/hip_runtime.h>
#include <hip/hip_bf16.h>
#include <math.h>

typedef __bf16 bf16_t;
typedef __bf16 v16bf __attribute__((ext_vector_type(16)));
typedef float  v8f   __attribute__((ext_vector_type(8)));

#define HIDC 256
#define LANE (threadIdx.x & 31)
#define WAVE (threadIdx.x >> 5)

__device__ __forceinline__ v8f wmma_bf16(v16bf a, v16bf b, v8f c) {
  // D = A(16x32) * B(32x16) + C, f32 accumulate
  return __builtin_amdgcn_wmma_f32_16x16x32_bf16(false, a, false, b, (short)0, c, false, false);
}

// A-operand fragment packing (16x32 bf16 tile, ISA 7.12.2):
// lane = m + 16*((k>>3)&1), half j = (k&7) + ((k>>4)<<3); lane holds 16 contiguous halves
__device__ __forceinline__ int apack_idx(int m, int kk) {
  return ((m + (((kk >> 3) & 1) << 4)) << 4) + ((kk & 7) + ((kk >> 4) << 3));
}
// B-operand fragment packing (32x16 bf16 tile): lane = n + 16*((k>>3)&1), half j = (k&7) + ((k>>4)<<3)
__device__ __forceinline__ int bpack_idx(int n, int kk) {
  return ((n + (((kk >> 3) & 1) << 4)) << 4) + ((kk & 7) + ((kk >> 4) << 3));
}
__device__ __forceinline__ float gelu_exact(float x) {
  return 0.5f * x * (1.0f + erff(x * 0.70710678118654752f));
}

// ---------------------------------------------------------------------------
// Pack a row-major f32 weight [K, N] (zero-padded to Kpad) into WMMA
// B-fragment tiles: tile(kt,nt) is 512 bf16 = 1024B; lane's 16 halves contiguous.
__global__ __launch_bounds__(256) void pack_b_kernel(const float* __restrict__ W,
                                                     bf16_t* __restrict__ out,
                                                     int K, int Kpad, int N) {
  int idx = blockIdx.x * 256 + threadIdx.x;
  if (idx >= Kpad * N) return;
  int k = idx / N, n = idx - k * N;
  float v = (k < K) ? W[(size_t)k * N + n] : 0.0f;
  int kt = k >> 5, kk = k & 31, nt = n >> 4, nn = n & 15;
  int ntiles = N >> 4;
  out[(size_t)(kt * ntiles + nt) * 512 + bpack_idx(nn, kk)] = (bf16_t)v;
}

// ---------------------------------------------------------------------------
// LayerNorm(x) -> bf16, and zero the scatter accumulators. One block per row.
__global__ __launch_bounds__(256) void ln_zero_kernel(const float* __restrict__ x,
                                                      const float* __restrict__ g,
                                                      const float* __restrict__ b,
                                                      bf16_t* __restrict__ xnbf,
                                                      float* __restrict__ sums,
                                                      float* __restrict__ cnt) {
  __shared__ float rs[8], rs2[8], mv[2];
  int row = blockIdx.x, t = threadIdx.x;
  float v = x[(size_t)row * HIDC + t];
  float s = v, s2 = v * v;
#pragma unroll
  for (int o = 16; o; o >>= 1) { s += __shfl_down(s, o, 32); s2 += __shfl_down(s2, o, 32); }
  if (LANE == 0) { rs[WAVE] = s; rs2[WAVE] = s2; }
  __syncthreads();
  if (t == 0) {
    float ts = 0.f, ts2 = 0.f;
    for (int i = 0; i < 8; ++i) { ts += rs[i]; ts2 += rs2[i]; }
    float m = ts * (1.0f / HIDC);
    mv[0] = m;
    mv[1] = rsqrtf(ts2 * (1.0f / HIDC) - m * m + 1e-5f);
  }
  __syncthreads();
  float xn = (v - mv[0]) * mv[1] * g[t] + b[t];
  xnbf[(size_t)row * HIDC + t] = (bf16_t)xn;
  sums[(size_t)row * HIDC + t] = 0.0f;
  if (t == 0) cnt[row] = 0.0f;
}

// ---------------------------------------------------------------------------
// Fused per-edge pipeline, 32 edges (2 M-tiles) per workgroup so every
// B-fragment load from L2 feeds TWO WMMAs. Dynamic LDS (~126 KB) — legal on
// CDNA5 (320 KB LDS per workgroup), far beyond the 64 KB static limit.
// LDS carve (bytes):
//   combA : 0      .. 34816   [2][17*512] bf16  comb A-fragments (K=544)
//   hidA  : 34816  .. 67584   [2][16*512] bf16  hidden A-fragments (K=512)
//   qkv   : 67584  .. 116736  [3][32*256] bf16
//   sc    : 116736 .. 124928  [32][8][8] f32    scores then attn
//   sbias : 124928 .. 125952  [32][8] f32
//   dstn  : 125952 .. 126080  [32] int
#define EDGE_SMEM 126080
#define EDGE_M 32

__global__ __launch_bounds__(256) void edge_kernel(
    const bf16_t* __restrict__ xnbf, const int* __restrict__ ei,
    const float* __restrict__ ea,
    const bf16_t* __restrict__ qW1p, const float* __restrict__ qb1,
    const bf16_t* __restrict__ qW2p, const float* __restrict__ qb2,
    const bf16_t* __restrict__ kW1p, const float* __restrict__ kb1,
    const bf16_t* __restrict__ kW2p, const float* __restrict__ kb2,
    const bf16_t* __restrict__ vW1p, const float* __restrict__ vb1,
    const bf16_t* __restrict__ vW2p, const float* __restrict__ vb2,
    const float* __restrict__ sW1, const float* __restrict__ sb1,
    const float* __restrict__ sW2, const float* __restrict__ sb2,
    float* __restrict__ sums, float* __restrict__ cnt, int E) {
  extern __shared__ __align__(32) char smem[];
  bf16_t* combA = (bf16_t*)smem;                  // [2][17*512]
  bf16_t* hidA  = (bf16_t*)(smem + 34816);        // [2][16*512]
  bf16_t* qkv   = (bf16_t*)(smem + 67584);        // [3][32*256]
  float*  sc    = (float*)(smem + 116736);        // [32][8][8]
  float*  sbias = (float*)(smem + 124928);        // [32][8]
  int*    dstn  = (int*)(smem + 125952);          // [32]

  const int e0 = blockIdx.x * EDGE_M;
  const int tid = threadIdx.x;

  // Gather comb = [xn[src], xn[dst], ea[:3], zeros] into A-fragment layout.
  for (int idx = tid; idx < EDGE_M * 544; idx += 256) {
    int m = idx / 544, c = idx - m * 544;
    int e = e0 + m;
    float v;
    if (c < 256)      v = (float)xnbf[(size_t)ei[e] * HIDC + c];
    else if (c < 512) v = (float)xnbf[(size_t)ei[E + e] * HIDC + (c - 256)];
    else if (c < 515) v = ea[(size_t)e * 7 + (c - 512)];
    else              v = 0.0f;
    combA[(m >> 4) * (17 * 512) + (c >> 5) * 512 + apack_idx(m & 15, c & 31)] = (bf16_t)v;
  }
  if (tid < EDGE_M) dstn[tid] = ei[E + e0 + tid];
  __syncthreads();

  const bf16_t* W1s[3] = {qW1p, kW1p, vW1p};
  const float*  b1s[3] = {qb1, kb1, vb1};
  const bf16_t* W2s[3] = {qW2p, kW2p, vW2p};
  const float*  b2s[3] = {qb2, kb2, vb2};

#pragma unroll
  for (int w = 0; w < 3; ++w) {
    // GEMM1: [32,544] x [544,512]; each B tile load feeds 2 WMMAs (2 M-tiles)
    for (int nt = WAVE; nt < 32; nt += 8) {
      v8f acc0 = {}, acc1 = {};
      for (int kt = 0; kt < 17; ++kt) {
        if (kt + 1 < 17)
          __builtin_prefetch(&W1s[w][(size_t)((kt + 1) * 32 + nt) * 512 + LANE * 16], 0, 3);
        v16bf bm = *(const v16bf*)&W1s[w][(size_t)(kt * 32 + nt) * 512 + LANE * 16];
        v16bf a0 = *(const v16bf*)&combA[kt * 512 + LANE * 16];
        v16bf a1 = *(const v16bf*)&combA[17 * 512 + kt * 512 + LANE * 16];
        acc0 = wmma_bf16(a0, bm, acc0);
        acc1 = wmma_bf16(a1, bm, acc1);
      }
      int ncol = nt * 16 + (LANE & 15);
      float bias1 = b1s[w][ncol];
      int kt2 = ncol >> 5, kk2 = ncol & 31;
#pragma unroll
      for (int r = 0; r < 8; ++r) {
        int m = r + ((LANE >> 4) << 3);
        hidA[kt2 * 512 + apack_idx(m, kk2)] = (bf16_t)gelu_exact(acc0[r] + bias1);
        hidA[16 * 512 + kt2 * 512 + apack_idx(m, kk2)] = (bf16_t)gelu_exact(acc1[r] + bias1);
      }
    }
    __syncthreads();
    // GEMM2: [32,512] x [512,256]
    for (int nt = WAVE; nt < 16; nt += 8) {
      v8f acc0 = {}, acc1 = {};
      for (int kt = 0; kt < 16; ++kt) {
        v16bf bm = *(const v16bf*)&W2s[w][(size_t)(kt * 16 + nt) * 512 + LANE * 16];
        v16bf a0 = *(const v16bf*)&hidA[kt * 512 + LANE * 16];
        v16bf a1 = *(const v16bf*)&hidA[16 * 512 + kt * 512 + LANE * 16];
        acc0 = wmma_bf16(a0, bm, acc0);
        acc1 = wmma_bf16(a1, bm, acc1);
      }
      int ncol = nt * 16 + (LANE & 15);
      float bias2 = b2s[w][ncol];
#pragma unroll
      for (int r = 0; r < 8; ++r) {
        int m = r + ((LANE >> 4) << 3);
        qkv[w * (EDGE_M * HIDC) + m * HIDC + ncol] = (bf16_t)(acc0[r] + bias2);
        qkv[w * (EDGE_M * HIDC) + (m + 16) * HIDC + ncol] = (bf16_t)(acc1[r] + bias2);
      }
    }
    __syncthreads();
  }

  // scores[e,h,g] = Q[e,h,:].K[e,g,:] / sqrt(32)
  for (int it = tid; it < EDGE_M * 64; it += 256) {
    int e = it >> 6, hg = it & 63, h = hg >> 3, g = hg & 7;
    float s = 0.f;
#pragma unroll 8
    for (int d = 0; d < 32; ++d)
      s += (float)qkv[e * HIDC + h * 32 + d] *
           (float)qkv[EDGE_M * HIDC + e * HIDC + g * 32 + d];
    sc[e * 64 + h * 8 + g] = s * 0.17677669529663687f;
  }
  // bias MLP: relu(ea4 @ sW1 + sb1) @ sW2 + sb2 -> [32,8]; one thread each
  {
    int e = tid >> 3, o = tid & 7;
    float a0 = ea[(size_t)(e0 + e) * 7 + 3], a1 = ea[(size_t)(e0 + e) * 7 + 4];
    float a2 = ea[(size_t)(e0 + e) * 7 + 5], a3 = ea[(size_t)(e0 + e) * 7 + 6];
    float acc = sb2[o];
    for (int h = 0; h < 64; ++h) {
      float t = sb1[h] + a0 * sW1[h] + a1 * sW1[64 + h] + a2 * sW1[128 + h] + a3 * sW1[192 + h];
      acc += fmaxf(t, 0.0f) * sW2[h * 8 + o];
    }
    sbias[e * 8 + o] = acc;
  }
  __syncthreads();
  // softmax over h (axis=1) per (e, g); one thread per column
  {
    int e = tid >> 3, g = tid & 7;
    float vals[8], mx = -1e30f;
#pragma unroll
    for (int h = 0; h < 8; ++h) {
      vals[h] = sc[e * 64 + h * 8 + g] + sbias[e * 8 + h];
      mx = fmaxf(mx, vals[h]);
    }
    float ssum = 0.f;
#pragma unroll
    for (int h = 0; h < 8; ++h) { vals[h] = expf(vals[h] - mx); ssum += vals[h]; }
    float inv = 1.0f / ssum;
#pragma unroll
    for (int h = 0; h < 8; ++h) sc[e * 64 + h * 8 + g] = vals[h] * inv;
  }
  __syncthreads();
  // wv[e,h,d] = sum_g attn[e,h,g] * V[e,g,d]; 8 threads per edge (one head each)
  {
    int e = tid >> 3, h = tid & 7;
    int node = dstn[e];
    const bf16_t* Ve = &qkv[2 * (EDGE_M * HIDC) + e * HIDC];
    const float* att = &sc[e * 64 + h * 8];
    for (int d = 0; d < 32; ++d) {
      float v = 0.f;
#pragma unroll
      for (int g = 0; g < 8; ++g) v += att[g] * (float)Ve[g * 32 + d];
      atomicAdd(&sums[(size_t)node * HIDC + h * 32 + d], v);
    }
    if (h == 0) atomicAdd(&cnt[node], 1.0f);
  }
}

// ---------------------------------------------------------------------------
// Node epilogue: mean, residual, rW GEMM, LN, final MLP, residual. 16 nodes/WG.
__global__ __launch_bounds__(256) void node_kernel(
    const float* __restrict__ x, const float* __restrict__ sums,
    const float* __restrict__ cnt,
    const bf16_t* __restrict__ rWp, const float* __restrict__ rb,
    const float* __restrict__ ln_g, const float* __restrict__ ln_b,
    const bf16_t* __restrict__ fW1p, const float* __restrict__ fb1,
    const bf16_t* __restrict__ fW2p, const float* __restrict__ fb2,
    float* __restrict__ out) {
  __shared__ float xbuf[16 * HIDC];                // x1 then x2
  __shared__ __align__(32) bf16_t catA[16 * 512];  // [x1, x] fragments, later LN fragments
  __shared__ __align__(32) bf16_t hidA[16 * 512];
  __shared__ float mu[16], rsd[16];

  const int n0 = blockIdx.x * 16;
  const int tid = threadIdx.x;

  // x1 = x + sums/max(cnt,1); build A-fragments of concat([x1, x])
  for (int idx = tid; idx < 16 * HIDC; idx += 256) {
    int m = idx >> 8, c = idx & 255;
    size_t g = (size_t)(n0 + m) * HIDC + c;
    float ct = fmaxf(cnt[n0 + m], 1.0f);
    float xv = x[g];
    float x1 = xv + sums[g] / ct;
    xbuf[m * HIDC + c] = x1;
    catA[(c >> 5) * 512 + apack_idx(m, c & 31)] = (bf16_t)x1;
    int c2 = c + 256;
    catA[(c2 >> 5) * 512 + apack_idx(m, c2 & 31)] = (bf16_t)xv;
  }
  __syncthreads();
  // x2 = x1 + [x1,x] @ rW + rb
  for (int nt = WAVE; nt < 16; nt += 8) {
    v8f acc = {};
    for (int kt = 0; kt < 16; ++kt) {
      v16bf a = *(const v16bf*)&catA[kt * 512 + LANE * 16];
      v16bf bm = *(const v16bf*)&rWp[(size_t)(kt * 16 + nt) * 512 + LANE * 16];
      acc = wmma_bf16(a, bm, acc);
    }
    int ncol = nt * 16 + (LANE & 15);
    float bb = rb[ncol];
#pragma unroll
    for (int r = 0; r < 8; ++r) {
      int m = r + ((LANE >> 4) << 3);
      xbuf[m * HIDC + ncol] += acc[r] + bb;   // now holds x2
    }
  }
  __syncthreads();
  // LN(x2) stats per row
  if (tid < 16) {
    float s = 0.f, s2 = 0.f;
    for (int c = 0; c < HIDC; ++c) { float v = xbuf[tid * HIDC + c]; s += v; s2 += v * v; }
    float m = s * (1.0f / HIDC);
    mu[tid] = m;
    rsd[tid] = rsqrtf(s2 * (1.0f / HIDC) - m * m + 1e-5f);
  }
  __syncthreads();
  // LN fragments (K=256 -> 8 k-tiles), reuse catA
  for (int idx = tid; idx < 16 * HIDC; idx += 256) {
    int m = idx >> 8, c = idx & 255;
    float v = (xbuf[m * HIDC + c] - mu[m]) * rsd[m] * ln_g[c] + ln_b[c];
    catA[(c >> 5) * 512 + apack_idx(m, c & 31)] = (bf16_t)v;
  }
  __syncthreads();
  // GEMM fW1 (256->512) + gelu
  for (int nt = WAVE; nt < 32; nt += 8) {
    v8f acc = {};
    for (int kt = 0; kt < 8; ++kt) {
      v16bf a = *(const v16bf*)&catA[kt * 512 + LANE * 16];
      v16bf bm = *(const v16bf*)&fW1p[(size_t)(kt * 32 + nt) * 512 + LANE * 16];
      acc = wmma_bf16(a, bm, acc);
    }
    int ncol = nt * 16 + (LANE & 15);
    float bb = fb1[ncol];
#pragma unroll
    for (int r = 0; r < 8; ++r) {
      float h = gelu_exact(acc[r] + bb);
      int m = r + ((LANE >> 4) << 3);
      hidA[(ncol >> 5) * 512 + apack_idx(m, ncol & 31)] = (bf16_t)h;
    }
  }
  __syncthreads();
  // GEMM fW2 (512->256) + x2 residual -> out
  for (int nt = WAVE; nt < 16; nt += 8) {
    v8f acc = {};
    for (int kt = 0; kt < 16; ++kt) {
      v16bf a = *(const v16bf*)&hidA[kt * 512 + LANE * 16];
      v16bf bm = *(const v16bf*)&fW2p[(size_t)(kt * 16 + nt) * 512 + LANE * 16];
      acc = wmma_bf16(a, bm, acc);
    }
    int ncol = nt * 16 + (LANE & 15);
    float bb = fb2[ncol];
#pragma unroll
    for (int r = 0; r < 8; ++r) {
      int m = r + ((LANE >> 4) << 3);
      out[(size_t)(n0 + m) * HIDC + ncol] = xbuf[m * HIDC + ncol] + acc[r] + bb;
    }
  }
}

// ---------------------------------------------------------------------------
extern "C" void kernel_launch(void* const* d_in, const int* in_sizes, int n_in,
                              void* d_out, int out_size, void* d_ws, size_t ws_size,
                              hipStream_t stream) {
  const float* x   = (const float*)d_in[0];
  const int*   ei  = (const int*)d_in[1];
  const float* ea  = (const float*)d_in[2];
  const float* qW1 = (const float*)d_in[3];  const float* qb1 = (const float*)d_in[4];
  const float* qW2 = (const float*)d_in[5];  const float* qb2 = (const float*)d_in[6];
  const float* kW1 = (const float*)d_in[7];  const float* kb1 = (const float*)d_in[8];
  const float* kW2 = (const float*)d_in[9];  const float* kb2 = (const float*)d_in[10];
  const float* vW1 = (const float*)d_in[11]; const float* vb1 = (const float*)d_in[12];
  const float* vW2 = (const float*)d_in[13]; const float* vb2 = (const float*)d_in[14];
  const float* sW1 = (const float*)d_in[15]; const float* sb1 = (const float*)d_in[16];
  const float* sW2 = (const float*)d_in[17]; const float* sb2 = (const float*)d_in[18];
  const float* lng = (const float*)d_in[19]; const float* lnb = (const float*)d_in[20];
  const float* rW  = (const float*)d_in[21]; const float* rb  = (const float*)d_in[22];
  const float* fW1 = (const float*)d_in[23]; const float* fb1 = (const float*)d_in[24];
  const float* fW2 = (const float*)d_in[25]; const float* fb2 = (const float*)d_in[26];
  float* out = (float*)d_out;

  const int N = in_sizes[0] / HIDC;   // 65536
  const int E = in_sizes[1] / 2;      // 262144

  char* w = (char*)d_ws;
  auto carve = [&](size_t bytes) {
    char* p = w;
    w += (bytes + 255) & ~(size_t)255;
    return p;
  };
  bf16_t* xnbf = (bf16_t*)carve((size_t)N * HIDC * 2);
  float*  sums = (float*)carve((size_t)N * HIDC * 4);
  float*  cnt  = (float*)carve((size_t)N * 4);
  const size_t W1P = (size_t)17 * 32 * 512 * 2;  // 544x512 packed
  const size_t W2P = (size_t)16 * 16 * 512 * 2;  // 512x256 packed
  bf16_t* qW1p = (bf16_t*)carve(W1P);
  bf16_t* kW1p = (bf16_t*)carve(W1P);
  bf16_t* vW1p = (bf16_t*)carve(W1P);
  bf16_t* qW2p = (bf16_t*)carve(W2P);
  bf16_t* kW2p = (bf16_t*)carve(W2P);
  bf16_t* vW2p = (bf16_t*)carve(W2P);
  bf16_t* rWp  = (bf16_t*)carve(W2P);                       // 512x256
  bf16_t* fW1p = (bf16_t*)carve((size_t)8 * 32 * 512 * 2);  // 256x512
  bf16_t* fW2p = (bf16_t*)carve(W2P);                       // 512x256

  auto pack = [&](const float* Wm, bf16_t* o, int K, int Kpad, int Nc) {
    int total = Kpad * Nc;
    pack_b_kernel<<<(total + 255) / 256, 256, 0, stream>>>(Wm, o, K, Kpad, Nc);
  };
  pack(qW1, qW1p, 515, 544, 512);
  pack(kW1, kW1p, 515, 544, 512);
  pack(vW1, vW1p, 515, 544, 512);
  pack(qW2, qW2p, 512, 512, 256);
  pack(kW2, kW2p, 512, 512, 256);
  pack(vW2, vW2p, 512, 512, 256);
  pack(rW,  rWp,  512, 512, 256);
  pack(fW1, fW1p, 256, 256, 512);
  pack(fW2, fW2p, 512, 512, 256);

  ln_zero_kernel<<<N, 256, 0, stream>>>(x, lng, lnb, xnbf, sums, cnt);

  edge_kernel<<<E / EDGE_M, 256, EDGE_SMEM, stream>>>(
      xnbf, ei, ea,
      qW1p, qb1, qW2p, qb2,
      kW1p, kb1, kW2p, kb2,
      vW1p, vb1, vW2p, vb2,
      sW1, sb1, sW2, sb2,
      sums, cnt, E);

  node_kernel<<<N / 16, 256, 0, stream>>>(
      x, sums, cnt, rWp, rb, lng, lnb, fW1p, fb1, fW2p, fb2, out);
}
